// BahdanauAttention_61538291417914
// MI455X (gfx1250) — matrix operand order repaired and checked by reading, not compiled, not run on hardware
//
#include <hip/hip_runtime.h>

typedef __attribute__((ext_vector_type(16))) _Float16 v16h;
typedef __attribute__((ext_vector_type(8)))  _Float16 v8h;
typedef __attribute__((ext_vector_type(8)))  float    v8f;

#define B_ 32
#define S_ 2048
#define H_ 1024
#define U_ 1024

// Branch-free tanh: hardware v_tanh_f32 if the builtin exists, else a
// single-v_exp_f32 rational form (saturates correctly to +/-1).
__device__ __forceinline__ float fast_tanh(float x) {
#if __has_builtin(__builtin_amdgcn_tanhf)
  return __builtin_amdgcn_tanhf(x);
#else
  return fmaf(-2.f, __frcp_rn(__expf(2.f * x) + 1.f), 1.f);
#endif
}

// ---------------------------------------------------------------------------
// Kernel 1: W1 [H,U] f32 (row-major) -> W1T [U,H] f16  (LDS-tiled transpose)
// B-fragments for v_wmma want contiguous K (=h) per output column u.
// ---------------------------------------------------------------------------
__global__ void __launch_bounds__(256)
k_w1t(const float* __restrict__ W1, _Float16* __restrict__ W1T) {
  __shared__ float tile[32][33];
  int u0 = (blockIdx.x & (U_ / 32 - 1)) * 32;
  int h0 = (blockIdx.x / (U_ / 32)) * 32;
  int tx = threadIdx.x & 31, ty = threadIdx.x >> 5;  // 32 x 8 threads, 4 passes
#pragma unroll
  for (int p = 0; p < 4; ++p)
    tile[ty + p * 8][tx] = W1[(size_t)(h0 + ty + p * 8) * U_ + u0 + tx];
  __syncthreads();
#pragma unroll
  for (int p = 0; p < 4; ++p)
    W1T[(size_t)(u0 + ty + p * 8) * H_ + h0 + tx] = (_Float16)tile[tx][ty + p * 8];
}

// ---------------------------------------------------------------------------
// Kernel 2: qproj[b,u] = query[b,:] @ W2[:,u] + b1[u] + b2[u]
// (b1 folded in here so the WMMA epilogue adds a single bias vector)
// ---------------------------------------------------------------------------
__global__ void __launch_bounds__(256)
k_qproj(const float* __restrict__ query, const float* __restrict__ W2,
        const float* __restrict__ b1, const float* __restrict__ b2,
        float* __restrict__ qproj) {
  int b = blockIdx.x >> 2;
  int u = ((blockIdx.x & 3) << 8) + threadIdx.x;
  float acc = b1[u] + b2[u];
  const float* q = query + (size_t)b * H_;
  const float* w = W2 + u;
#pragma unroll 4
  for (int h = 0; h < H_; ++h)
    acc = fmaf(q[h], w[(size_t)h * U_], acc);   // q[h] uniform -> scalar load
  qproj[(size_t)b * U_ + u] = acc;
}

// ---------------------------------------------------------------------------
// Kernel 3 (WMMA): per (b, 16-row s-tile) fuse
//   score[m] = sum_u tanh( (values_tile @ W1)[m,u] + qproj[b,u] ) * V[u]
// 8 waves/block, wave owns u-tiles {w, w+8, ...}, 32 k-steps of
// v_wmma_f32_16x16x32_f16 per u-tile.
// ---------------------------------------------------------------------------
__global__ void __launch_bounds__(256)
k_scores(const float* __restrict__ values, const _Float16* __restrict__ W1T,
         const float* __restrict__ qproj, const float* __restrict__ V,
         float* __restrict__ scores) {
  __shared__ _Float16 sval[16 * H_];   // 32 KB f16 A-tile
  __shared__ float    sqp[U_];         // 4 KB bias row
  __shared__ float    sV[U_];          // 4 KB V vector
  __shared__ float    ssc[16];         // per-row score accumulators

  int tid = threadIdx.x;
  int b  = blockIdx.x >> 7;            // S_/16 = 128 tiles per batch
  int s0 = (blockIdx.x & 127) << 4;

  // Stage the contiguous 16x1024 f32 tile, converting to f16 in LDS.
  const float* vsrc = values + ((size_t)b * S_ + s0) * H_;
  for (int i = tid * 4; i < 16 * H_; i += 256 * 4) {
    float4 f = *(const float4*)(vsrc + i);
    sval[i + 0] = (_Float16)f.x;  sval[i + 1] = (_Float16)f.y;
    sval[i + 2] = (_Float16)f.z;  sval[i + 3] = (_Float16)f.w;
  }
  for (int i = tid; i < U_; i += 256) {
    sqp[i] = qproj[(size_t)b * U_ + i];
    sV[i]  = V[i];
  }
  if (tid < 16) ssc[tid] = 0.f;
  __syncthreads();

  int wave = tid >> 5, lane = tid & 31;
  int half = lane >> 4;                // K-half selector per ISA A/B layout
  int n    = lane & 15;                // A-row (M) and B-column (N)

  float part[8];
#pragma unroll
  for (int j = 0; j < 8; ++j) part[j] = 0.f;

  // A-fragment: lane holds row M=n, 16 contiguous K starting at half*16
  const _Float16* arow = sval + n * H_ + half * 16;

  for (int ut = wave; ut < U_ / 16; ut += 8) {
    int u0 = ut << 4;
    // B-fragment: lane holds column u0+n, 16 contiguous K starting at half*16
    const _Float16* bp = W1T + (size_t)(u0 + n) * H_ + half * 16;
    v8f acc = {};
#pragma unroll 4
    for (int kb = 0; kb < H_; kb += 32) {
      union { v16h v; v8h h[2]; } A;
      A.h[0] = *(const v8h*)(arow + kb);        // 16B LDS load
      A.h[1] = *(const v8h*)(arow + kb + 8);    // 16B LDS load
      v16h Bv = *(const v16h*)(bp + kb);        // 32B global load (L2-hot)
      acc = __builtin_amdgcn_wmma_f32_16x16x32_f16(
          false, A.v, false, Bv, (short)0, acc, false, false);
    }
    // Epilogue on C layout: VGPR j -> row m = j + half*8, col = u0+n
    float qpn = sqp[u0 + n];
    float vn  = sV[u0 + n];
#pragma unroll
    for (int j = 0; j < 8; ++j) {
      float t = fast_tanh(acc[j] + qpn);        // branchless TRANS op
      part[j] = fmaf(t, vn, part[j]);
    }
  }
  // Reduce across the 16 N-lanes and the 8 waves via ds_add_f32.
#pragma unroll
  for (int j = 0; j < 8; ++j)
    atomicAdd(&ssc[j + half * 8], part[j]);
  __syncthreads();
  if (tid < 16) scores[(size_t)b * S_ + s0 + tid] = ssc[tid];
}

// ---------------------------------------------------------------------------
// Kernel 4: softmax over S per batch (bv dropped: softmax shift-invariant)
// ---------------------------------------------------------------------------
__global__ void __launch_bounds__(256)
k_softmax(const float* __restrict__ scores, float* __restrict__ attn) {
  __shared__ float red[256];
  int b = blockIdx.x, tid = threadIdx.x;
  const float* sc = scores + (size_t)b * S_;
  float m = -1e30f;
  for (int s = tid; s < S_; s += 256) m = fmaxf(m, sc[s]);
  red[tid] = m; __syncthreads();
  for (int o = 128; o > 0; o >>= 1) {
    if (tid < o) red[tid] = fmaxf(red[tid], red[tid + o]);
    __syncthreads();
  }
  m = red[0]; __syncthreads();
  float sum = 0.f;
  for (int s = tid; s < S_; s += 256) sum += __expf(sc[s] - m);
  red[tid] = sum; __syncthreads();
  for (int o = 128; o > 0; o >>= 1) {
    if (tid < o) red[tid] += red[tid + o];
    __syncthreads();
  }
  float inv = 1.f / red[0];
  for (int s = tid; s < S_; s += 256)
    attn[(size_t)b * S_ + s] = __expf(sc[s] - m) * inv;
}

// ---------------------------------------------------------------------------
// Kernel 5a: zero the context output (atomics accumulate into it)
// ---------------------------------------------------------------------------
__global__ void k_zero(float* __restrict__ p) {
  p[(size_t)blockIdx.x * 256 + threadIdx.x] = 0.f;
}

// ---------------------------------------------------------------------------
// Kernel 5b: context[b,h] = sum_s attn[b,s] * values[b,s,h]
// S split across blocks for occupancy; f32 global atomics to combine.
// ---------------------------------------------------------------------------
__global__ void __launch_bounds__(256)
k_context(const float* __restrict__ values, const float* __restrict__ attn,
          float* __restrict__ ctx) {
  int b = blockIdx.x;
  int h = (blockIdx.y << 8) + threadIdx.x;
  int sbeg = blockIdx.z << 8;
  const float* w  = attn + (size_t)b * S_ + sbeg;
  const float* vp = values + ((size_t)b * S_ + sbeg) * H_ + h;
  float acc = 0.f;
#pragma unroll 4
  for (int s = 0; s < 256; ++s)
    acc = fmaf(w[s], vp[(size_t)s * H_], acc);  // w[s] uniform -> scalar load
  atomicAdd(&ctx[(size_t)b * H_ + h], acc);
}

// ---------------------------------------------------------------------------
extern "C" void kernel_launch(void* const* d_in, const int* in_sizes, int n_in,
                              void* d_out, int out_size, void* d_ws, size_t ws_size,
                              hipStream_t stream) {
  const float* query  = (const float*)d_in[0];
  const float* values = (const float*)d_in[1];
  const float* W1     = (const float*)d_in[2];
  const float* b1     = (const float*)d_in[3];
  const float* W2     = (const float*)d_in[4];
  const float* b2     = (const float*)d_in[5];
  const float* V      = (const float*)d_in[6];
  // d_in[7] = bv: constant shift before softmax -> no effect on outputs.

  char* ws = (char*)d_ws;
  _Float16* W1T  = (_Float16*)ws;                                   // 2 MB
  float*    qprj = (float*)(ws + (size_t)U_ * H_ * 2);              // 128 KB
  float*    scor = (float*)(ws + (size_t)U_ * H_ * 2 + (size_t)B_ * U_ * 4); // 256 KB

  float* ctx  = (float*)d_out;            // [B,H]   (first output, flat)
  float* attn = (float*)d_out + B_ * H_;  // [B,S,1] (second output, flat)

  k_w1t    <<<(U_ / 32) * (H_ / 32), 256, 0, stream>>>(W1, W1T);
  k_qproj  <<<B_ * (U_ / 256),       256, 0, stream>>>(query, W2, b1, b2, qprj);
  k_scores <<<B_ * (S_ / 16),        256, 0, stream>>>(values, W1T, qprj, V, scor);
  k_softmax<<<B_,                    256, 0, stream>>>(scor, attn);
  k_zero   <<<(B_ * H_) / 256,       256, 0, stream>>>(ctx);
  k_context<<<dim3(B_, H_ / 256, S_ / 256), 256, 0, stream>>>(values, attn, ctx);
}
